// FourierBlock_54717883351622
// MI455X (gfx1250) — matrix operand order repaired and checked by reading, not compiled, not run on hardware
//
#include <hip/hip_runtime.h>
#include <hip/hip_bf16.h>

// ---------------------------------------------------------------------------
// FourierBlock for MI455X (gfx1250, wave32, WMMA bf16 16x16x32, f32 accum)
//
// rfft folded into W' = F @ sum_p(w); irfft folded into Vir = vx_re@C+vx_im@S.
// Pipeline (intermediates in d_ws, bf16 unless noted):
//   K1  wsum   = sum_p w[p]                      (fp32 complex, per bank)
//   K2  W'T    = (F @ wsum)^T  [6][544][1024]
//   K3  CST    = irfft cos/sin basis^T [1024][1088]
//   K4  xbf    = bf16(x)       [32768][1024]
//   K5  qxr/qxi/kxr/kxi [32768][544], vxcat [32768][1088]   (6 WMMA GEMMs)
//   K5b virT   = per-batch (vxcat @ CST)^T  [32][1024][1024] (WMMA GEMM)
//   K6  attention: scores(8 wmma/pass) -> |.| -> softmax -> a @ virT -> out
// ---------------------------------------------------------------------------

typedef __attribute__((ext_vector_type(16))) __bf16 v16bf;
typedef __attribute__((ext_vector_type(8)))  __bf16 v8bf;
typedef __attribute__((ext_vector_type(8)))  float  v8f;

#define B_SZ   32
#define LLEN   1024
#define FOU    513
#define NPAD   544          // 513 padded to multiple of 32
#define NPAD2  1088         // re||im concat
#define MROWS  32768        // B*H
#define TWO_PI 6.28318530717958647692f

__device__ __forceinline__ v8f wmma_bf16(v16bf a, v16bf b, v8f c) {
  return __builtin_amdgcn_wmma_f32_16x16x32_bf16(false, a, false, b, (short)0, c,
                                                 false, false);
}

// A fragment (16x32, M x K): lane m+16*hl holds K = [8hl,8hl+8) u [16+8hl,+8)
__device__ __forceinline__ v16bf ldfragA(const __bf16* p) {
  v8bf lo = *(const v8bf*)(p);
  v8bf hi = *(const v8bf*)(p + 16);
  v16bf r;
#pragma unroll
  for (int i = 0; i < 8; ++i) { r[i] = lo[i]; r[i + 8] = hi[i]; }
  return r;
}

// B fragment (32x16, K x N) read from B^T [N,K]: lane n+16*hl holds K = [8hl,8hl+8) u [16+8hl,+8)
__device__ __forceinline__ v16bf ldfragB(const __bf16* p) {
  v8bf lo = *(const v8bf*)(p);
  v8bf hi = *(const v8bf*)(p + 16);
  v16bf r;
#pragma unroll
  for (int i = 0; i < 8; ++i) { r[i] = lo[i]; r[i + 8] = hi[i]; }
  return r;
}

// --------------------------- K1: sum weight banks ---------------------------
__global__ void k_sum_w(const float2* __restrict__ wq, const float2* __restrict__ wk,
                        const float2* __restrict__ wv, float2* __restrict__ wsum) {
  const size_t NN = (size_t)FOU * FOU;
  size_t idx = (size_t)blockIdx.x * blockDim.x + threadIdx.x;
  if (idx >= 3 * NN) return;
  int bank = (int)(idx / NN);
  size_t e  = idx % NN;
  const float2* w = (bank == 0) ? wq : (bank == 1) ? wk : wv;
  float2 a = w[e], b = w[NN + e], c = w[2 * NN + e];
  float2 r; r.x = a.x + b.x + c.x; r.y = a.y + b.y + c.y;
  wsum[idx] = r;
}

// ------------------- K2: W'T[bank][o][l] = (F @ wsum)^T ---------------------
// F[l][i] = exp(-2*pi*i*l*i/L) built by rotation recurrence (1 sincos/thread).
__global__ void k_build_wt(const float2* __restrict__ wsum, __bf16* __restrict__ dst) {
  size_t idx = (size_t)blockIdx.x * blockDim.x + threadIdx.x;
  if (idx >= (size_t)3 * NPAD * LLEN) return;
  int bank = (int)(idx / ((size_t)NPAD * LLEN));
  int rem  = (int)(idx % ((size_t)NPAD * LLEN));
  int o = rem / LLEN, l = rem % LLEN;
  float accr = 0.f, acci = 0.f;
  if (o < FOU) {
    float sn, cs;
    __sincosf(-TWO_PI * (float)l / (float)LLEN, &sn, &cs);
    float c = 1.f, s = 0.f;
    const float2* W = wsum + (size_t)bank * FOU * FOU;
    for (int i = 0; i < FOU; ++i) {
      float2 w = W[(size_t)i * FOU + o];
      accr += c * w.x - s * w.y;
      acci += c * w.y + s * w.x;
      float c2 = c * cs - s * sn;
      s = s * cs + c * sn;
      c = c2;
    }
  }
  dst[(size_t)(2 * bank + 0) * NPAD * LLEN + (size_t)o * LLEN + l] = (__bf16)accr;
  dst[(size_t)(2 * bank + 1) * NPAD * LLEN + (size_t)o * LLEN + l] = (__bf16)acci;
}

// ---------------- K3: CST[l][f] = irfft basis, transposed -------------------
__global__ void k_build_cst(__bf16* __restrict__ cst) {
  int idx = blockIdx.x * blockDim.x + threadIdx.x;
  if (idx >= LLEN * NPAD2) return;
  int l = idx / NPAD2, f = idx % NPAD2;
  float v = 0.f;
  if (f < FOU) {
    float coef = ((f == 0 || f == FOU - 1) ? 1.f : 2.f) / (float)LLEN;
    v = coef * __cosf(TWO_PI * (float)f * (float)l / (float)LLEN);
  } else if (f >= NPAD && f < NPAD + FOU) {
    int fi = f - NPAD;
    float coef = ((fi == 0 || fi == FOU - 1) ? 1.f : 2.f) / (float)LLEN;
    v = -coef * __sinf(TWO_PI * (float)fi * (float)l / (float)LLEN);
  }
  cst[idx] = (__bf16)v;
}

// --------------------------- K4: x -> bf16 ---------------------------------
__global__ void k_x2bf(const float* __restrict__ x, __bf16* __restrict__ xbf, size_t n) {
  size_t i = (size_t)blockIdx.x * blockDim.x + threadIdx.x;
  if (i < n) xbf[i] = (__bf16)x[i];
}

// ------------------ K5: WMMA GEMM, 4 N-tiles per wave -----------------------
// C[M,N] = A[M,K] @ B[K,N], B supplied transposed (BT[N,K]).
// Per K-step: issue A-frag + 4 B-frag loads as one group, then 4 WMMAs.
// mode 0: C row-major [M,ldc] bf16, column offset col0.
// mode 1: per-batch transposed out: row = b*1024+g -> C[b][n][g]  (for virT)
__global__ void __launch_bounds__(128)
k_gemm_bf16(const __bf16* __restrict__ A, int lda,
            const __bf16* __restrict__ BT, int ldb,
            __bf16* __restrict__ C, int ldc, int col0,
            int ksteps, int ntiles, int mode) {
  const int lane = threadIdx.x & 31;
  const int wv   = threadIdx.x >> 5;            // 4 waves
  const int mt   = blockIdx.x;
  const int ntg  = (blockIdx.y * 4 + wv) * 4;   // first of this wave's 4 N-tiles
  if (ntg >= ntiles) return;
  const int n  = lane & 15;   // M-row for A frag, N-col for B frag / C
  const int hl = lane >> 4;

  const __bf16* a0 = A + (size_t)(mt * 16 + n) * lda + 8 * hl;
  const __bf16* b0[4];
#pragma unroll
  for (int t = 0; t < 4; ++t) {
    int nt = ntg + t;
    if (nt >= ntiles) nt = ntg;                 // clamp: computed, never stored
    b0[t] = BT + (size_t)(nt * 16 + n) * ldb + 8 * hl;
  }

  v8f acc[4] = {};
  for (int ks = 0; ks < ksteps; ++ks) {
    __builtin_prefetch(a0 + ks * 32 + 256, 0, 0);   // 8 K-steps ahead, speculative
    // issue all loads for this K-step before any WMMA consumes them
    v16bf af = ldfragA(a0 + ks * 32);
    v16bf bf[4];
#pragma unroll
    for (int t = 0; t < 4; ++t) bf[t] = ldfragB(b0[t] + ks * 32);
#pragma unroll
    for (int t = 0; t < 4; ++t) acc[t] = wmma_bf16(af, bf[t], acc[t]);
  }

#pragma unroll
  for (int t = 0; t < 4; ++t) {
    int nt = ntg + t;
    if (nt >= ntiles) continue;
    if (mode == 0) {
      size_t base = (size_t)(mt * 16 + 8 * hl) * ldc + col0 + nt * 16 + n;
#pragma unroll
      for (int r = 0; r < 8; ++r) C[base + (size_t)r * ldc] = (__bf16)acc[t][r];
    } else {
#pragma unroll
      for (int r = 0; r < 8; ++r) {
        int grow = mt * 16 + r + 8 * hl;
        int b = grow >> 10, g = grow & 1023;
        C[((size_t)b << 20) + (size_t)(nt * 16 + n) * 1024 + g] = (__bf16)acc[t][r];
      }
    }
  }
}

// -------------------- K6: fused attention (per 16-row tile) -----------------
// grid = (64 h-tiles, 32 batches), block = 256 (8 waves), 96KB dynamic LDS.
__global__ void __launch_bounds__(256)
k_attn(const __bf16* __restrict__ qxr, const __bf16* __restrict__ qxi,
       const __bf16* __restrict__ kxr, const __bf16* __restrict__ kxi,
       const __bf16* __restrict__ virT, float* __restrict__ out) {
  extern __shared__ char smem[];
  float*  sc = (float*)smem;                          // [16][1024] scores
  __bf16* ab = (__bf16*)(smem + 16 * 1024 * 4);       // [16][1024] softmaxed bf16

  const int b    = blockIdx.y;
  const int ht   = blockIdx.x;
  const int lane = threadIdx.x & 31;
  const int wv   = threadIdx.x >> 5;
  const int n    = lane & 15;
  const int hl   = lane >> 4;

  const size_t rowQ = (size_t)b * 1024 + ht * 16 + n;
  const __bf16* qra = qxr + rowQ * NPAD + 8 * hl;
  const __bf16* qia = qxi + rowQ * NPAD + 8 * hl;

  // ---- phase 1: complex scores qk = qx @ kx^T, two g-tiles per pass ----
  // 6 loads issued as a group, then 8 WMMAs per K-step.
  for (int gp = wv; gp < 32; gp += 8) {
    const int gt0 = gp * 2;
    const size_t rowK = (size_t)b * 1024 + gt0 * 16 + n;
    const __bf16* kr0 = kxr + rowK * NPAD + 8 * hl;
    const __bf16* ki0 = kxi + rowK * NPAD + 8 * hl;
    const __bf16* kr1 = kr0 + (size_t)16 * NPAD;
    const __bf16* ki1 = ki0 + (size_t)16 * NPAD;
    v8f rr0 = {}, ii0 = {}, ri0 = {}, ir0 = {};
    v8f rr1 = {}, ii1 = {}, ri1 = {}, ir1 = {};
    for (int ks = 0; ks < NPAD / 32; ++ks) {
      v16bf aqr = ldfragA(qra + ks * 32);
      v16bf aqi = ldfragA(qia + ks * 32);
      v16bf b0r = ldfragB(kr0 + ks * 32);
      v16bf b0i = ldfragB(ki0 + ks * 32);
      v16bf b1r = ldfragB(kr1 + ks * 32);
      v16bf b1i = ldfragB(ki1 + ks * 32);
      rr0 = wmma_bf16(aqr, b0r, rr0);
      ii0 = wmma_bf16(aqi, b0i, ii0);
      ri0 = wmma_bf16(aqr, b0i, ri0);
      ir0 = wmma_bf16(aqi, b0r, ir0);
      rr1 = wmma_bf16(aqr, b1r, rr1);
      ii1 = wmma_bf16(aqi, b1i, ii1);
      ri1 = wmma_bf16(aqr, b1i, ri1);
      ir1 = wmma_bf16(aqi, b1r, ir1);
    }
#pragma unroll
    for (int r = 0; r < 8; ++r) {
      float re0 = rr0[r] - ii0[r], im0 = ri0[r] + ir0[r];
      float re1 = rr1[r] - ii1[r], im1 = ri1[r] + ir1[r];
      sc[(r + 8 * hl) * 1024 + gt0 * 16 + n]       = __fsqrt_rn(re0 * re0 + im0 * im0);
      sc[(r + 8 * hl) * 1024 + (gt0 + 1) * 16 + n] = __fsqrt_rn(re1 * re1 + im1 * im1);
    }
  }
  __syncthreads();

  // ---- phase 2: row softmax (2 rows per wave), write bf16 'a' to LDS ----
  for (int rr2 = 0; rr2 < 2; ++rr2) {
    int row = wv * 2 + rr2;
    float m = -1e30f;
    for (int g = lane; g < 1024; g += 32) m = fmaxf(m, sc[row * 1024 + g]);
#pragma unroll
    for (int off = 16; off > 0; off >>= 1) m = fmaxf(m, __shfl_xor(m, off, 32));
    float s = 0.f;
    for (int g = lane; g < 1024; g += 32) s += __expf(sc[row * 1024 + g] - m);
#pragma unroll
    for (int off = 16; off > 0; off >>= 1) s += __shfl_xor(s, off, 32);
    float inv = 1.f / s;
    for (int g = lane; g < 1024; g += 32)
      ab[row * 1024 + g] = (__bf16)(__expf(sc[row * 1024 + g] - m) * inv);
  }
  __syncthreads();

  // ---- phase 3: out = a @ virT^T, 4 N-tiles per pass (1 LDS A-frag -> 4 WMMA) ----
  const __bf16* av = ab + n * 1024 + 8 * hl;  // A frags from LDS
  for (int g4 = 0; g4 < 2; ++g4) {
    const int nt0 = wv * 8 + g4 * 4;
    const __bf16* bv[4];
#pragma unroll
    for (int t = 0; t < 4; ++t)
      bv[t] = virT + ((size_t)b << 20) + (size_t)((nt0 + t) * 16 + n) * 1024 + 8 * hl;
    v8f acc[4] = {};
    for (int ks = 0; ks < 32; ++ks) {
      v16bf af = ldfragA(av + ks * 32);
      v16bf bf[4];
#pragma unroll
      for (int t = 0; t < 4; ++t) bf[t] = ldfragB(bv[t] + ks * 32);
#pragma unroll
      for (int t = 0; t < 4; ++t) acc[t] = wmma_bf16(af, bf[t], acc[t]);
    }
#pragma unroll
    for (int t = 0; t < 4; ++t)
#pragma unroll
      for (int r = 0; r < 8; ++r)
        out[((size_t)b << 20) + (size_t)(ht * 16 + r + 8 * hl) * 1024 +
            (nt0 + t) * 16 + n] = acc[t][r];
  }
}

// ---------------------------------------------------------------------------
extern "C" void kernel_launch(void* const* d_in, const int* in_sizes, int n_in,
                              void* d_out, int out_size, void* d_ws, size_t ws_size,
                              hipStream_t stream) {
  const float*  x  = (const float*)d_in[0];
  const float2* wk = (const float2*)d_in[1];   // dict order: x, w_k, w_q, w_v
  const float2* wq = (const float2*)d_in[2];
  const float2* wv = (const float2*)d_in[3];
  float* out = (float*)d_out;

  // workspace partition (~363 MB total)
  size_t off = 0;
  auto alloc = [&](size_t bytes) -> char* {
    char* p = (char*)d_ws + off;
    off += (bytes + 255) & ~(size_t)255;
    return p;
  };
  __bf16* xbf  = (__bf16*)alloc((size_t)MROWS * 1024 * 2);
  float2* wsum = (float2*)alloc((size_t)3 * FOU * FOU * 8);
  __bf16* wt   = (__bf16*)alloc((size_t)6 * NPAD * LLEN * 2);   // qr,qi,kr,ki,vr,vi
  __bf16* cst  = (__bf16*)alloc((size_t)LLEN * NPAD2 * 2);
  __bf16* qxr  = (__bf16*)alloc((size_t)MROWS * NPAD * 2);
  __bf16* qxi  = (__bf16*)alloc((size_t)MROWS * NPAD * 2);
  __bf16* kxr_ = (__bf16*)alloc((size_t)MROWS * NPAD * 2);
  __bf16* kxi_ = (__bf16*)alloc((size_t)MROWS * NPAD * 2);
  __bf16* vxc  = (__bf16*)alloc((size_t)MROWS * NPAD2 * 2);
  __bf16* virT = (__bf16*)alloc((size_t)B_SZ * 1024 * 1024 * 2);
  (void)ws_size; (void)in_sizes; (void)n_in; (void)out_size;

  const size_t WTS = (size_t)NPAD * LLEN;  // one W'T matrix

  // weight prep
  {
    size_t n = (size_t)3 * FOU * FOU;
    k_sum_w<<<dim3((unsigned)((n + 255) / 256)), dim3(256), 0, stream>>>(wq, wk, wv, wsum);
  }
  {
    size_t n = (size_t)3 * NPAD * LLEN;
    k_build_wt<<<dim3((unsigned)((n + 255) / 256)), dim3(256), 0, stream>>>(wsum, wt);
  }
  k_build_cst<<<dim3((LLEN * NPAD2) / 256), dim3(256), 0, stream>>>(cst);
  {
    size_t n = (size_t)MROWS * 1024;
    k_x2bf<<<dim3((unsigned)((n + 255) / 256)), dim3(256), 0, stream>>>(x, xbf, n);
  }

  // stage 1: x @ W'  (6 real GEMMs, M=32768 K=1024 N=544)
  // block = 128 (4 waves), 16 N-tiles per block -> grid.y = ceil(34/16) = 3
  const dim3 g1(MROWS / 16, 3), blk(128);
  k_gemm_bf16<<<g1, blk, 0, stream>>>(xbf, 1024, wt + 0 * WTS, 1024, qxr, NPAD, 0,   32, 34, 0);
  k_gemm_bf16<<<g1, blk, 0, stream>>>(xbf, 1024, wt + 1 * WTS, 1024, qxi, NPAD, 0,   32, 34, 0);
  k_gemm_bf16<<<g1, blk, 0, stream>>>(xbf, 1024, wt + 2 * WTS, 1024, kxr_, NPAD, 0,  32, 34, 0);
  k_gemm_bf16<<<g1, blk, 0, stream>>>(xbf, 1024, wt + 3 * WTS, 1024, kxi_, NPAD, 0,  32, 34, 0);
  k_gemm_bf16<<<g1, blk, 0, stream>>>(xbf, 1024, wt + 4 * WTS, 1024, vxc, NPAD2, 0,    32, 34, 0);
  k_gemm_bf16<<<g1, blk, 0, stream>>>(xbf, 1024, wt + 5 * WTS, 1024, vxc, NPAD2, NPAD, 32, 34, 0);

  // stage 1.5: virT[b][l][g] = ((vx_re@C + vx_im@S))^T  (M=32768 K=1088 N=1024)
  k_gemm_bf16<<<dim3(MROWS / 16, 4), blk, 0, stream>>>(vxc, NPAD2, cst, NPAD2,
                                                       virT, 0, 0, 34, 64, 1);

  // stage 2: fused scores -> |.| -> softmax -> a @ virT -> out
  const size_t lds = 16 * 1024 * 4 + 16 * 1024 * 2;  // 96 KB
  k_attn<<<dim3(64, B_SZ), dim3(256), lds, stream>>>(qxr, qxi, kxr_, kxi_, virT, out);
}
